// Speller_3779571220513
// MI455X (gfx1250) — matrix operand order repaired and checked by reading, not compiled, not run on hardware
//
#include <hip/hip_runtime.h>
#include <hip/hip_bf16.h>
#include <math.h>

// ---------------------------------------------------------------------------
// Problem constants (from reference)
// ---------------------------------------------------------------------------
#define B_   64
#define TENC 2000
#define TDEC 250
#define V_   31
#define EMB_ 256
#define HID_ 256
#define SP_  512
#define LH_  512
#define G4   2048      // 4*SP
#define XK   1024      // concat [input(512) || h(512)]

// ---------------------------------------------------------------------------
// CDNA5 async global->LDS path (guarded: falls back to plain staging if the
// toolchain does not declare the gfx1250 async builtins).
// Probe feedback: the builtin takes pointers to 16-byte int vectors, global
// source in AS1, LDS destination in AS3.
// ---------------------------------------------------------------------------
#if __has_builtin(__builtin_amdgcn_global_load_async_to_lds_b128) && \
    __has_builtin(__builtin_amdgcn_s_wait_asynccnt)
#define USE_ASYNC_LDS 1
typedef int v4i_t __attribute__((vector_size(4 * sizeof(int))));
typedef __attribute__((address_space(1))) v4i_t gas_v4i;
typedef __attribute__((address_space(3))) v4i_t las_v4i;
#else
#define USE_ASYNC_LDS 0
#endif

// ---------------------------------------------------------------------------
// WMMA fragment types (CDNA5 gfx1250, wave32)
// ---------------------------------------------------------------------------
typedef __attribute__((ext_vector_type(16))) __bf16 v16bf;
typedef __attribute__((ext_vector_type(8)))  float  v8f;

union ABfrag {
    v16bf  v;
    __bf16 h[16];
    uint4  u[2];
};

// A-matrix 16x32 bf16 per ISA layout:
//   lanes 0-15  : row M = m0+lane,     K = {k0..k0+7, k0+16..k0+23}
//   lanes 16-31 : row M = m0+lane-16,  K = {k0+8..k0+15, k0+24..k0+31}
__device__ __forceinline__ void load_a_bf16(const __bf16* A, int lda, int m0,
                                            int k0, int lane, ABfrag& a) {
    int m  = m0 + (lane & 15);
    int kk = k0 + ((lane >> 4) << 3);
    const __bf16* p = A + (size_t)m * lda + kk;
    a.u[0] = *(const uint4*)(p);
    a.u[1] = *(const uint4*)(p + 16);
}

// B-matrix 32x16 bf16, B[k][n] = W[n][k] (computing X @ W.T, W row-major [N,K]):
//   lanes 0-15  : col N = n0+lane,     K = {k0..k0+7, k0+16..k0+23}
//   lanes 16-31 : col N = n0+lane-16,  K = {k0+8..k0+15, k0+24..k0+31}
__device__ __forceinline__ void load_b_bf16(const __bf16* W, int ldw, int n0,
                                            int k0, int lane, ABfrag& b) {
    int n  = n0 + (lane & 15);
    int kk = k0 + ((lane >> 4) << 3);
    const __bf16* p = W + (size_t)n * ldw + kk;
    b.u[0] = *(const uint4*)(p);
    b.u[1] = *(const uint4*)(p + 16);
}

__device__ __forceinline__ float sigf(float x) { return 1.0f / (1.0f + __expf(-x)); }

// ---------------------------------------------------------------------------
// One-time: zero the recurrent-state region of the workspace
// ---------------------------------------------------------------------------
__global__ void zero_ws_kernel(unsigned int* p, size_t n_words) {
    size_t i = (size_t)blockIdx.x * blockDim.x + threadIdx.x;
    if (i < n_words) p[i] = 0u;
}

// ---------------------------------------------------------------------------
// One-time: prepack LSTM weights to bf16 [w_ih || w_hh] -> [2048,1024],
// bias sums, and qw to bf16.
// ---------------------------------------------------------------------------
__global__ void prepack_kernel(const float* wih0, const float* whh0,
                               const float* bih0, const float* bhh0,
                               const float* wih1, const float* whh1,
                               const float* bih1, const float* bhh1,
                               const float* wih2, const float* whh2,
                               const float* bih2, const float* bhh2,
                               const float* qww,
                               __bf16* wc0, __bf16* wc1, __bf16* wc2,
                               float* bs0, float* bs1, float* bs2,
                               __bf16* qwb) {
    size_t idx = (size_t)blockIdx.x * blockDim.x + threadIdx.x;
    const size_t per_cell = (size_t)G4 * XK;
    if (idx < 3 * per_cell) {
        int cell = (int)(idx / per_cell);
        size_t rem = idx % per_cell;
        int n = (int)(rem >> 10);
        int k = (int)(rem & (XK - 1));
        const float* wih = (cell == 0) ? wih0 : (cell == 1) ? wih1 : wih2;
        const float* whh = (cell == 0) ? whh0 : (cell == 1) ? whh1 : whh2;
        float v = (k < SP_) ? wih[(size_t)n * SP_ + k]
                            : whh[(size_t)n * SP_ + (k - SP_)];
        __bf16* wc = (cell == 0) ? wc0 : (cell == 1) ? wc1 : wc2;
        wc[rem] = (__bf16)v;
    }
    if (idx < 3 * G4) {
        int cell = (int)(idx / G4);
        int n = (int)(idx % G4);
        const float* bih = (cell == 0) ? bih0 : (cell == 1) ? bih1 : bih2;
        const float* bhh = (cell == 0) ? bhh0 : (cell == 1) ? bhh1 : bhh2;
        float* bs = (cell == 0) ? bs0 : (cell == 1) ? bs1 : bs2;
        bs[n] = bih[n] + bhh[n];
    }
    if (idx < (size_t)HID_ * SP_) {
        qwb[idx] = (__bf16)qww[idx];
    }
}

// ---------------------------------------------------------------------------
// One-time: K = enc @ kw.T + kw_b  (and V likewise, blockIdx.y selects).
// Output stored as bf16 so K+V (131 MB) stays resident in the 192 MB L2.
// A (enc) is fp32 in memory -> converted to bf16 fragments in registers.
// ---------------------------------------------------------------------------
__global__ void kv_gemm_kernel(const float* __restrict__ enc,
                               const float* __restrict__ kw, const float* __restrict__ kb,
                               const float* __restrict__ vw, const float* __restrict__ vb,
                               __bf16* __restrict__ Kout, __bf16* __restrict__ Vout) {
    const float* W  = (blockIdx.y == 0) ? kw : vw;
    const float* Bs = (blockIdx.y == 0) ? kb : vb;
    __bf16* Out     = (blockIdx.y == 0) ? Kout : Vout;

    int lane = threadIdx.x & 31;
    size_t wave = (((size_t)blockIdx.x * blockDim.x) + threadIdx.x) >> 5; // 0..127999
    int tn = (int)(wave & 15);          // 16 n-tiles (N = 256)
    size_t tm = wave >> 4;              // 8000 m-tiles (M = 128000 rows)
    size_t m0 = tm * 16;
    int n0 = tn * 16;

    v8f c = {};
    for (int k0 = 0; k0 < LH_; k0 += 32) {
        ABfrag a, bf;
        {   // A fragment: fp32 -> bf16 in registers
            size_t m = m0 + (lane & 15);
            int kk = k0 + ((lane >> 4) << 3);
            const float* pa = enc + m * LH_ + kk;
#pragma unroll
            for (int j = 0; j < 8; ++j) {
                a.h[j]     = (__bf16)pa[j];
                a.h[8 + j] = (__bf16)pa[16 + j];
            }
        }
        {   // B fragment: W row n, contiguous K slice, fp32 -> bf16
            int n = n0 + (lane & 15);
            int kk = k0 + ((lane >> 4) << 3);
            const float* pw = W + (size_t)n * LH_ + kk;
#pragma unroll
            for (int j = 0; j < 8; ++j) {
                bf.h[j]     = (__bf16)pw[j];
                bf.h[8 + j] = (__bf16)pw[16 + j];
            }
        }
        c = __builtin_amdgcn_wmma_f32_16x16x32_bf16(false, a.v, false, bf.v,
                                                    (short)0, c, false, false);
    }
    int n = n0 + (lane & 15);
    float bias = Bs[n];
    int mb = (lane >> 4) << 3;
#pragma unroll
    for (int r = 0; r < 8; ++r) {
        size_t m = m0 + r + mb;
        Out[m * HID_ + n] = (__bf16)(c[r] + bias);
    }
}

// ---------------------------------------------------------------------------
// Per-step: write embedding of input token into xh0[:, 0:256] (bf16)
// ---------------------------------------------------------------------------
__global__ void build_emb_kernel(const int* __restrict__ y, const float* __restrict__ E,
                                 __bf16* __restrict__ xh0, int t) {
    int b = blockIdx.x, k = threadIdx.x;            // 64 x 256
    int tok = (t == 0) ? 1 : y[b * TDEC + (t - 1)]; // SOS = 1
    xh0[(size_t)b * XK + k] = (__bf16)E[(size_t)tok * EMB_ + k];
}

// ---------------------------------------------------------------------------
// Per-step per-cell: g = [x||h] @ [w_ih||w_hh].T   (M=64, N=2048, K=1024)
// One wave = one 16x16 tile; 512 tiles -> 64 blocks of 8 waves.
// All 8 waves of a block share the same 16 A-rows -> stage them once into
// LDS via the CDNA5 async global->LDS path (ASYNCcnt), read via ds_load.
// ---------------------------------------------------------------------------
__global__ void gates_gemm_kernel(const __bf16* __restrict__ X,
                                  const __bf16* __restrict__ Wc,
                                  float* __restrict__ g) {
    __shared__ __bf16 xs[16 * XK];                 // 32 KB of 320 KB/WGP
    int wave = (blockIdx.x * blockDim.x + threadIdx.x) >> 5;  // 0..511
    int tm = wave >> 7;                            // constant within a block
    int m0 = tm * 16;

    // ---- stage X rows [m0, m0+16) into LDS, 16B chunks, 8 per thread ----
    {
        const __bf16* src = X + (size_t)m0 * XK;
#if USE_ASYNC_LDS
        for (int i = threadIdx.x; i < (16 * XK) / 8; i += 256) {
            __builtin_amdgcn_global_load_async_to_lds_b128(
                (gas_v4i*)(src + (size_t)i * 8),
                (las_v4i*)(xs + (size_t)i * 8),
                0, 0);
        }
        __builtin_amdgcn_s_wait_asynccnt(0);
#else
        for (int i = threadIdx.x; i < (16 * XK) / 8; i += 256)
            *(uint4*)(xs + (size_t)i * 8) = *(const uint4*)(src + (size_t)i * 8);
#endif
        __syncthreads();
    }

    int lane = threadIdx.x & 31;
    int tn = wave & 127;      // 128 n-tiles
    int n0 = tn * 16;

    v8f c = {};
    for (int k0 = 0; k0 < XK; k0 += 32) {
        ABfrag a, bf;
        {   // A fragment from LDS (local row = lane&15)
            int mr = lane & 15;
            int kk = k0 + ((lane >> 4) << 3);
            const __bf16* p = xs + (size_t)mr * XK + kk;
            a.u[0] = *(const uint4*)(p);
            a.u[1] = *(const uint4*)(p + 16);
        }
        load_b_bf16(Wc, XK, n0, k0, lane, bf);
        if (k0 + 32 < XK) {  // prefetch next weight tile -> global_prefetch_b8
            int n = n0 + (lane & 15);
            __builtin_prefetch(Wc + (size_t)n * XK + k0 + 32, 0, 0);
        }
        c = __builtin_amdgcn_wmma_f32_16x16x32_bf16(false, a.v, false, bf.v,
                                                    (short)0, c, false, false);
    }
    int n = n0 + (lane & 15);
    int mb = (lane >> 4) << 3;
#pragma unroll
    for (int r = 0; r < 8; ++r)
        g[(size_t)(m0 + r + mb) * G4 + n] = c[r];
}

// ---------------------------------------------------------------------------
// Per-step per-cell: pointwise LSTM update. Writes bf16 h to two concat
// destinations (feeding this cell's recurrence and the next cell's input),
// optionally fp32 h (cell 2, needed by the output head).
// ---------------------------------------------------------------------------
__global__ void lstm_pointwise_kernel(const float* __restrict__ g,
                                      const float* __restrict__ bs,
                                      float* __restrict__ cst,
                                      __bf16* __restrict__ d1, int off1,
                                      __bf16* __restrict__ d2, int off2, int stride2,
                                      float* __restrict__ hf) {
    int idx = blockIdx.x * blockDim.x + threadIdx.x;  // 0..32767
    int b = idx >> 9, j = idx & 511;
    const float* gr = g + (size_t)b * G4;
    float iv = sigf(gr[j]            + bs[j]);
    float fv = sigf(gr[SP_ + j]      + bs[SP_ + j]);
    float gv = tanhf(gr[2 * SP_ + j] + bs[2 * SP_ + j]);
    float ov = sigf(gr[3 * SP_ + j]  + bs[3 * SP_ + j]);
    float c = fv * cst[idx] + iv * gv;
    cst[idx] = c;
    float h = ov * tanhf(c);
    __bf16 hb = (__bf16)h;
    d1[(size_t)b * XK + off1 + j] = hb;
    d2[(size_t)b * stride2 + off2 + j] = hb;
    if (hf) hf[idx] = h;
}

// ---------------------------------------------------------------------------
// Per-step: q = h2 @ qw.T + qw_b   (M=64, N=256, K=512) via WMMA
// ---------------------------------------------------------------------------
__global__ void q_gemm_kernel(const __bf16* __restrict__ h2b,
                              const __bf16* __restrict__ qwb,
                              const float* __restrict__ qbias,
                              float* __restrict__ qf) {
    int lane = threadIdx.x & 31;
    int wave = (blockIdx.x * blockDim.x + threadIdx.x) >> 5;  // 0..63
    int tm = wave >> 4;      // 4 m-tiles
    int tn = wave & 15;      // 16 n-tiles
    int m0 = tm * 16, n0 = tn * 16;

    v8f c = {};
    for (int k0 = 0; k0 < SP_; k0 += 32) {
        ABfrag a, bf;
        load_a_bf16(h2b, SP_, m0, k0, lane, a);
        load_b_bf16(qwb, SP_, n0, k0, lane, bf);
        c = __builtin_amdgcn_wmma_f32_16x16x32_bf16(false, a.v, false, bf.v,
                                                    (short)0, c, false, false);
    }
    int n = n0 + (lane & 15);
    float bias = qbias[n];
    int mb = (lane >> 4) << 3;
#pragma unroll
    for (int r = 0; r < 8; ++r)
        qf[(size_t)(m0 + r + mb) * HID_ + n] = c[r] + bias;
}

// ---------------------------------------------------------------------------
// Per-step: raw[b,te] = (1/sqrt(HID)) * dot(K[b,te,:], q[b,:])
// K is bf16, L2-resident. One block = one (b, 256-te chunk).
// ---------------------------------------------------------------------------
__global__ void attn_scores_kernel(const __bf16* __restrict__ Kb,
                                   const float* __restrict__ qf,
                                   float* __restrict__ out_att, int t) {
    int b = blockIdx.y;
    int te = blockIdx.x * 256 + threadIdx.x;
    __shared__ float qs[HID_];
    qs[threadIdx.x] = qf[(size_t)b * HID_ + threadIdx.x];
    __syncthreads();
    if (te >= TENC) return;
    const __bf16* kr = Kb + ((size_t)b * TENC + te) * HID_;
    float acc = 0.0f;
#pragma unroll 4
    for (int k = 0; k < HID_; k += 8) {
        uint4 u = *(const uint4*)(kr + k);      // 8 bf16 = b128 load
        const __bf16* hp = (const __bf16*)&u;
#pragma unroll
        for (int j = 0; j < 8; ++j) acc += (float)hp[j] * qs[k + j];
    }
    out_att[((size_t)b * TDEC + t) * TENC + te] = acc * 0.0625f; // 1/sqrt(256)
}

// ---------------------------------------------------------------------------
// Per-step: softmax over Tenc=2000, in place in the output attention plot.
// ---------------------------------------------------------------------------
__global__ void softmax_kernel(float* __restrict__ out_att, int t) {
    int b = blockIdx.x;
    float* a = out_att + ((size_t)b * TDEC + t) * TENC;
    __shared__ float red[256];
    int tid = threadIdx.x;

    float mx = -INFINITY;
    for (int i = tid; i < TENC; i += 256) mx = fmaxf(mx, a[i]);
    red[tid] = mx; __syncthreads();
    for (int s = 128; s > 0; s >>= 1) {
        if (tid < s) red[tid] = fmaxf(red[tid], red[tid + s]);
        __syncthreads();
    }
    mx = red[0]; __syncthreads();

    float sum = 0.0f;
    for (int i = tid; i < TENC; i += 256) {
        float e = __expf(a[i] - mx);
        a[i] = e;
        sum += e;
    }
    red[tid] = sum; __syncthreads();
    for (int s = 128; s > 0; s >>= 1) {
        if (tid < s) red[tid] += red[tid + s];
        __syncthreads();
    }
    float inv = 1.0f / red[0];
    for (int i = tid; i < TENC; i += 256) a[i] *= inv;
}

// ---------------------------------------------------------------------------
// Per-step: ctx[b,k] = sum_te attw[b,te] * V[b,te,k]. Coalesced bf16 reads of
// L2-resident V. Writes fp32 ctx (head) and bf16 ctx into xh0[:, 256:512].
// ---------------------------------------------------------------------------
__global__ void ctx_kernel(const __bf16* __restrict__ Vb,
                           const float* __restrict__ out_att,
                           float* __restrict__ ctxf,
                           __bf16* __restrict__ xh0, int t) {
    int b = blockIdx.x, k = threadIdx.x;     // 64 x 256
    const float* a = out_att + ((size_t)b * TDEC + t) * TENC;
    const __bf16* vb = Vb + (size_t)b * TENC * HID_ + k;
    float acc = 0.0f;
#pragma unroll 4
    for (int te = 0; te < TENC; ++te)
        acc += a[te] * (float)vb[(size_t)te * HID_];
    ctxf[(size_t)b * HID_ + k] = acc;
    xh0[(size_t)b * XK + EMB_ + k] = (__bf16)acc;
}

// ---------------------------------------------------------------------------
// Per-step output head: o=tanh([h2||ctx]@oc.T+b); o=tanh(o@l1.T+b);
// pred = o@E.T + cp_b. Tiny (64x31), one block per batch row.
// ---------------------------------------------------------------------------
__global__ void head_kernel(const float* __restrict__ h2f,
                            const float* __restrict__ ctxf,
                            const float* __restrict__ ocw, const float* __restrict__ ocb,
                            const float* __restrict__ l1w, const float* __restrict__ l1b,
                            const float* __restrict__ E, const float* __restrict__ cpb,
                            float* __restrict__ out_raw, int t) {
    int b = blockIdx.x, tid = threadIdx.x;   // 64 x 256
    __shared__ float cdn[SP_ + HID_];
    __shared__ float o1[EMB_];
    __shared__ float o2[EMB_];
    cdn[tid]        = h2f[(size_t)b * SP_ + tid];
    cdn[256 + tid]  = h2f[(size_t)b * SP_ + 256 + tid];
    cdn[512 + tid]  = ctxf[(size_t)b * HID_ + tid];
    __syncthreads();

    float acc = ocb[tid];
    const float* wr = ocw + (size_t)tid * (SP_ + HID_);
    for (int k = 0; k < SP_ + HID_; ++k) acc += cdn[k] * wr[k];
    o1[tid] = tanhf(acc);
    __syncthreads();

    acc = l1b[tid];
    const float* lr = l1w + (size_t)tid * EMB_;
    for (int k = 0; k < EMB_; ++k) acc += o1[k] * lr[k];
    o2[tid] = tanhf(acc);
    __syncthreads();

    if (tid < V_) {
        acc = cpb[tid];
        const float* er = E + (size_t)tid * EMB_;
        for (int k = 0; k < EMB_; ++k) acc += o2[k] * er[k];
        out_raw[((size_t)b * TDEC + t) * V_ + tid] = acc;
    }
}

// ---------------------------------------------------------------------------
// Launch
// ---------------------------------------------------------------------------
extern "C" void kernel_launch(void* const* d_in, const int* in_sizes, int n_in,
                              void* d_out, int out_size, void* d_ws, size_t ws_size,
                              hipStream_t stream) {
    const float* enc   = (const float*)d_in[0];
    const int*   y     = (const int*)  d_in[1];
    const float* E     = (const float*)d_in[2];
    const float* kw_w  = (const float*)d_in[3];
    const float* kw_b  = (const float*)d_in[4];
    const float* vw_w  = (const float*)d_in[5];
    const float* vw_b  = (const float*)d_in[6];
    const float* qw_w  = (const float*)d_in[7];
    const float* qw_b  = (const float*)d_in[8];
    const float* w_ih0 = (const float*)d_in[9];
    const float* w_hh0 = (const float*)d_in[10];
    const float* b_ih0 = (const float*)d_in[11];
    const float* b_hh0 = (const float*)d_in[12];
    const float* w_ih1 = (const float*)d_in[13];
    const float* w_hh1 = (const float*)d_in[14];
    const float* b_ih1 = (const float*)d_in[15];
    const float* b_hh1 = (const float*)d_in[16];
    const float* w_ih2 = (const float*)d_in[17];
    const float* w_hh2 = (const float*)d_in[18];
    const float* b_ih2 = (const float*)d_in[19];
    const float* b_hh2 = (const float*)d_in[20];
    const float* oc_w  = (const float*)d_in[21];
    const float* oc_b  = (const float*)d_in[22];
    const float* l1_w  = (const float*)d_in[23];
    const float* l1_b  = (const float*)d_in[24];
    const float* cp_b  = (const float*)d_in[25];

    float* out_raw = (float*)d_out;
    float* out_att = out_raw + (size_t)B_ * TDEC * V_;

    // ------------------ workspace layout ------------------
    char* ws = (char*)d_ws;
    size_t off = 0;
    auto alloc = [&](size_t bytes) -> char* {
        char* p = ws + off;
        off = (off + bytes + 255) & ~(size_t)255;
        return p;
    };
    __bf16* Kb  = (__bf16*)alloc((size_t)B_ * TENC * HID_ * 2);
    __bf16* Vb  = (__bf16*)alloc((size_t)B_ * TENC * HID_ * 2);
    __bf16* wc0 = (__bf16*)alloc((size_t)G4 * XK * 2);
    __bf16* wc1 = (__bf16*)alloc((size_t)G4 * XK * 2);
    __bf16* wc2 = (__bf16*)alloc((size_t)G4 * XK * 2);
    float*  bs0 = (float*) alloc((size_t)G4 * 4);
    float*  bs1 = (float*) alloc((size_t)G4 * 4);
    float*  bs2 = (float*) alloc((size_t)G4 * 4);
    __bf16* qwb = (__bf16*)alloc((size_t)HID_ * SP_ * 2);

    size_t state_start = off;
    float*  c0   = (float*) alloc((size_t)B_ * SP_ * 4);
    float*  c1   = (float*) alloc((size_t)B_ * SP_ * 4);
    float*  c2   = (float*) alloc((size_t)B_ * SP_ * 4);
    float*  ctxf = (float*) alloc((size_t)B_ * HID_ * 4);
    float*  h2f  = (float*) alloc((size_t)B_ * SP_ * 4);
    __bf16* xh0  = (__bf16*)alloc((size_t)B_ * XK * 2);
    __bf16* xh1  = (__bf16*)alloc((size_t)B_ * XK * 2);
    __bf16* xh2  = (__bf16*)alloc((size_t)B_ * XK * 2);
    __bf16* h2b  = (__bf16*)alloc((size_t)B_ * SP_ * 2);
    float*  qf   = (float*) alloc((size_t)B_ * HID_ * 4);
    float*  g    = (float*) alloc((size_t)B_ * G4 * 4);
    size_t state_bytes = off - state_start;

    // ------------------ one-time setup ------------------
    {
        size_t words = state_bytes / 4;
        int blocks = (int)((words + 255) / 256);
        zero_ws_kernel<<<blocks, 256, 0, stream>>>((unsigned int*)(ws + state_start), words);
    }
    {
        size_t total = (size_t)3 * G4 * XK;
        int blocks = (int)((total + 255) / 256);
        prepack_kernel<<<blocks, 256, 0, stream>>>(
            w_ih0, w_hh0, b_ih0, b_hh0,
            w_ih1, w_hh1, b_ih1, b_hh1,
            w_ih2, w_hh2, b_ih2, b_hh2,
            qw_w, wc0, wc1, wc2, bs0, bs1, bs2, qwb);
    }
    // K/V projection: 8000 m-tiles x 16 n-tiles = 128000 waves, y selects K vs V
    kv_gemm_kernel<<<dim3(16000, 2), 256, 0, stream>>>(enc, kw_w, kw_b, vw_w, vw_b, Kb, Vb);

    // ------------------ sequential decode ------------------
    for (int t = 0; t < TDEC; ++t) {
        build_emb_kernel<<<B_, 256, 0, stream>>>(y, E, xh0, t);

        // cell 0: x = [emb || ctx], h -> xh0[:,512:]; also feeds xh1[:,0:512]
        gates_gemm_kernel<<<64, 256, 0, stream>>>(xh0, wc0, g);
        lstm_pointwise_kernel<<<128, 256, 0, stream>>>(g, bs0, c0,
                                                       xh0, SP_, xh1, 0, XK, nullptr);
        // cell 1
        gates_gemm_kernel<<<64, 256, 0, stream>>>(xh1, wc1, g);
        lstm_pointwise_kernel<<<128, 256, 0, stream>>>(g, bs1, c1,
                                                       xh1, SP_, xh2, 0, XK, nullptr);
        // cell 2 (h2 also to dedicated bf16 buffer + fp32 for the head)
        gates_gemm_kernel<<<64, 256, 0, stream>>>(xh2, wc2, g);
        lstm_pointwise_kernel<<<128, 256, 0, stream>>>(g, bs2, c2,
                                                       xh2, SP_, h2b, 0, SP_, h2f);

        q_gemm_kernel<<<8, 256, 0, stream>>>(h2b, qwb, qw_b, qf);
        attn_scores_kernel<<<dim3(8, B_), 256, 0, stream>>>(Kb, qf, out_att, t);
        softmax_kernel<<<B_, 256, 0, stream>>>(out_att, t);
        ctx_kernel<<<B_, 256, 0, stream>>>(Vb, out_att, ctxf, xh0, t);
        head_kernel<<<B_, 256, 0, stream>>>(h2f, ctxf, oc_w, oc_b, l1_w, l1_b,
                                            E, cp_b, out_raw, t);
    }
}
